// GRUMPNN_54949811585637
// MI455X (gfx1250) — matrix operand order repaired and checked by reading, not compiled, not run on hardware
//
#include <hip/hip_runtime.h>
#include <stdint.h>

typedef __attribute__((ext_vector_type(16))) __bf16 v16bf;
typedef __attribute__((ext_vector_type(8)))  float  v8f;

// ---------- helpers ----------

// A-fragment (16x32 bf16) per ISA layout; lane<16 -> K {0..7,16..23},
// lane>=16 -> K {8..15,24..31}. koff = (lane>>4)*8, c0 = chunk base col.
__device__ __forceinline__ v16bf load_a_frag(const float* __restrict__ row,
                                             int c0, int koff) {
    v16bf a;
#pragma unroll
    for (int j = 0; j < 8; ++j) {
        a[j]     = (__bf16)row[c0 + koff + j];
        a[8 + j] = (__bf16)row[c0 + 16 + koff + j];
    }
    return a;
}

__device__ __forceinline__ v8f wmma_bf16(v16bf a, const uint32_t* bptr, v8f c) {
    v16bf b = *reinterpret_cast<const v16bf*>(bptr);   // 32B-aligned lane slice (LDS)
    return __builtin_amdgcn_wmma_f32_16x16x32_bf16(false, a, false, b,
                                                   (short)0, c, false, false);
}

__device__ __forceinline__ v8f zero8() { v8f c = {0.f,0.f,0.f,0.f,0.f,0.f,0.f,0.f}; return c; }

__device__ __forceinline__ unsigned enc_f(float f) {
    unsigned u = __float_as_uint(f);
    return (u & 0x80000000u) ? ~u : (u | 0x80000000u);
}
__device__ __forceinline__ float dec_f(unsigned u) {
    u = (u & 0x80000000u) ? (u & 0x7FFFFFFFu) : ~u;
    return __uint_as_float(u);
}
__device__ __forceinline__ float sigmoidf_(float x) { return 1.0f / (1.0f + __expf(-x)); }
__device__ __forceinline__ float tanhf_(float x) {
    float t = __expf(2.0f * x);
    return 1.0f - 2.0f / (t + 1.0f);
}

// ---------- weight packing: W [Nout x K] row-major -> B-fragment tiles ----------
// Tile (kt,nt) = 32x16 bf16, index t = kt*NT + nt, 256 dwords/tile; lane's 8
// dwords contiguous at t*256 + lane*8: n = nt*16+(lane&15), Klocal {0..7,16..23} for
// lane<16 else {8..15,24..31} (ISA A-matching order), 2 per dword.
__global__ void pack_b_kernel(const float* __restrict__ W, int K, int NT,
                              uint32_t* __restrict__ dstp) {
    int id = blockIdx.x * blockDim.x + threadIdx.x;
    int total = (K / 32) * NT * 256;
    if (id >= total) return;
    int v    = id & 7;
    int lane = (id >> 3) & 31;
    int t    = id >> 8;
    int nt = t % NT;
    int kt = t / NT;
    int n  = nt * 16 + (lane & 15);
    int k0 = kt * 32 + ((lane < 16) ? 0 : 8) + ((v < 4) ? 2 * v : 16 + 2 * (v - 4));
    union { __bf16 h[2]; uint32_t u; } p;
    p.h[0] = (__bf16)W[(size_t)n * K + k0];
    p.h[1] = (__bf16)W[(size_t)n * K + k0 + 1];
    dstp[(size_t)t * 256 + lane * 8 + v] = p.u;
}

// ---------- init segment buffers ----------
__global__ void init_seg_kernel(unsigned* __restrict__ m_enc, float* __restrict__ s,
                                float* __restrict__ agg, int N) {
    int i = blockIdx.x * blockDim.x + threadIdx.x;
    if (i < N * 64) agg[i] = 0.0f;
    if (i < N) { m_enc[i] = 0u; s[i] = 0.0f; }
}

// LDS tile offsets (dwords) inside the staged edge weight block
#define EW_WIE 0        // 48 tiles (K=128 x N=192)
#define EW_WHE 12288    // 24 tiles (K=64  x N=192)
#define EW_W1  18432    // 24 tiles (K=192 x N=64)
#define EW_TOT 24576    // dwords (96 KB)
#define NW_WIN 0        // 24 tiles (K=64 x N=192)
#define NW_WHN 6144     // 24 tiles (K=64 x N=192)
#define NW_TOT 12288    // dwords (48 KB)

// ---------- edge kernel: GRU edge update + attention logits + segment max ----------
__global__ __launch_bounds__(256) void edge_kernel(
    const float* __restrict__ nf, const float* __restrict__ ef_in,
    const int* __restrict__ src, const int* __restrict__ dst,
    const uint32_t* __restrict__ pWedge,   // Wie | Whe | W1 packed, contiguous
    const float* __restrict__ bie, const float* __restrict__ bhe,
    const float* __restrict__ b1, const float* __restrict__ W2,
    const float* __restrict__ b2,
    float* __restrict__ ef_out, float* __restrict__ logit_out,
    unsigned* __restrict__ m_enc, int n_edges)
{
    extern __shared__ uint32_t lds_w[];
    // stage all edge weights into LDS once per workgroup (vectorized)
    {
        const uint4* sv = (const uint4*)pWedge;
        uint4* dv = (uint4*)lds_w;
        for (int i = threadIdx.x; i < EW_TOT / 4; i += blockDim.x) dv[i] = sv[i];
    }
    __syncthreads();

    int wave = blockIdx.x * (blockDim.x >> 5) + (threadIdx.x >> 5);
    int lane = threadIdx.x & 31;
    int base = wave * 16;
    if (base >= n_edges) return;

    int row   = lane & 15;
    int koff  = (lane >> 4) * 8;
    int e_row = min(base + row, n_edges - 1);
    const float* hs = nf + (size_t)src[e_row] * 64;
    const float* hd = nf + (size_t)dst[e_row] * 64;
    const float* eh = ef_in + (size_t)e_row * 64;

    // A fragments: x = [hs|hd] (K=128, 4 chunks), ef (K=64, 2 chunks)
    v16bf xA[4], eA[2];
    xA[0] = load_a_frag(hs, 0, koff);  xA[1] = load_a_frag(hs, 32, koff);
    xA[2] = load_a_frag(hd, 0, koff);  xA[3] = load_a_frag(hd, 32, koff);
    eA[0] = load_a_frag(eh, 0, koff);  eA[1] = load_a_frag(eh, 32, koff);

    int ncol     = lane & 15;
    int mrow_off = (lane < 16) ? 0 : 8;
    int boff     = lane * 8;

    // r (tiles 0..3) and z (tiles 4..7): zero-init C, fuse gi+gh, bias in activation
    v8f rz[8];
#pragma unroll
    for (int t = 0; t < 8; ++t) {
        int gn = t * 16 + ncol;
        float cinit = bie[gn] + bhe[gn];
        v8f c = zero8();
#pragma unroll
        for (int kt = 0; kt < 4; ++kt)
            c = wmma_bf16(xA[kt], lds_w + EW_WIE + (kt * 12 + t) * 256 + boff, c);
#pragma unroll
        for (int kt = 0; kt < 2; ++kt)
            c = wmma_bf16(eA[kt], lds_w + EW_WHE + (kt * 12 + t) * 256 + boff, c);
#pragma unroll
        for (int i = 0; i < 8; ++i) c[i] = sigmoidf_(c[i] + cinit);
        rz[t] = c;
    }

    // n gate (tiles 8..11): i_n and h_n kept separate; combine + write uef
#pragma unroll
    for (int t = 0; t < 4; ++t) {
        int nt = 8 + t;
        int gn = nt * 16 + ncol;
        float bi0 = bie[gn], bh0 = bhe[gn];
        v8f ci = zero8(), ch = zero8();
#pragma unroll
        for (int kt = 0; kt < 4; ++kt)
            ci = wmma_bf16(xA[kt], lds_w + EW_WIE + (kt * 12 + nt) * 256 + boff, ci);
#pragma unroll
        for (int kt = 0; kt < 2; ++kt)
            ch = wmma_bf16(eA[kt], lds_w + EW_WHE + (kt * 12 + nt) * 256 + boff, ch);
        v8f r = rz[t], z = rz[4 + t];
        int feat = t * 16 + ncol;
#pragma unroll
        for (int i = 0; i < 8; ++i) {
            int e2 = base + i + mrow_off;
            if (e2 < n_edges) {
                float nn = tanhf_((ci[i] + bi0) + r[i] * (ch[i] + bh0));
                float h  = ef_in[(size_t)e2 * 64 + feat];
                ef_out[(size_t)e2 * 64 + feat] = (1.0f - z[i]) * nn + z[i] * h;
            }
        }
    }

    // attention MLP: hidden = relu(am @ W1^T + b1); logit = hidden @ W2^T + b2
    float part[8];
#pragma unroll
    for (int i = 0; i < 8; ++i) part[i] = 0.0f;
#pragma unroll
    for (int t = 0; t < 4; ++t) {
        int hn = t * 16 + ncol;
        float b1v = b1[hn];
        v8f c = zero8();
#pragma unroll
        for (int kt = 0; kt < 4; ++kt)            // am cols 0..127 = x
            c = wmma_bf16(xA[kt], lds_w + EW_W1 + (kt * 4 + t) * 256 + boff, c);
#pragma unroll
        for (int kt = 0; kt < 2; ++kt)            // am cols 128..191 = ef
            c = wmma_bf16(eA[kt], lds_w + EW_W1 + ((4 + kt) * 4 + t) * 256 + boff, c);
        float w2v = W2[hn];
#pragma unroll
        for (int i = 0; i < 8; ++i) part[i] += fmaxf(c[i] + b1v, 0.0f) * w2v;
    }
    // reduce across the 16 lanes of each half (N dimension of C tiles)
#pragma unroll
    for (int m = 1; m <= 8; m <<= 1) {
#pragma unroll
        for (int i = 0; i < 8; ++i) part[i] += __shfl_xor(part[i], m, 32);
    }
    if (row < 8) {
        int e2 = base + row + mrow_off;
        if (e2 < n_edges) {
            float lg = part[row] + b2[0];
            logit_out[e2] = lg;
            atomicMax(&m_enc[dst[e2]], enc_f(lg));
        }
    }
}

// ---------- softmax numerator + segment sum ----------
__global__ void exp_sum_kernel(float* __restrict__ logit_e, const unsigned* __restrict__ m_enc,
                               const int* __restrict__ dst, float* __restrict__ s, int E) {
    int e = blockIdx.x * blockDim.x + threadIdx.x;
    if (e >= E) return;
    int d = dst[e];
    float ex = __expf(logit_e[e] - dec_f(m_enc[d]));
    logit_e[e] = ex;                 // in-place: logit buffer becomes exp buffer
    atomicAdd(&s[d], ex);
}

// ---------- attention-weighted scatter into node aggregate ----------
__global__ void agg_scatter_kernel(const float* __restrict__ uef, const float* __restrict__ ebuf,
                                   const float* __restrict__ s, const int* __restrict__ dst,
                                   float* __restrict__ agg, int E) {
    int idx = blockIdx.x * blockDim.x + threadIdx.x;
    if (idx >= E * 64) return;
    int e = idx >> 6, f = idx & 63;
    int d = dst[e];
    float attn = ebuf[e] / s[d];
    atomicAdd(&agg[(size_t)d * 64 + f], uef[idx] * attn);
}

// ---------- node kernel: GRU node update ----------
__global__ __launch_bounds__(256) void node_kernel(
    const float* __restrict__ agg, const float* __restrict__ nf_in,
    const uint32_t* __restrict__ pWnode,   // Win | Whn packed, contiguous
    const float* __restrict__ bin, const float* __restrict__ bhn,
    float* __restrict__ nf_out, int n_nodes)
{
    extern __shared__ uint32_t lds_w[];
    {
        const uint4* sv = (const uint4*)pWnode;
        uint4* dv = (uint4*)lds_w;
        for (int i = threadIdx.x; i < NW_TOT / 4; i += blockDim.x) dv[i] = sv[i];
    }
    __syncthreads();                      // barrier before any wave may exit

    int wave = blockIdx.x * (blockDim.x >> 5) + (threadIdx.x >> 5);
    int lane = threadIdx.x & 31;
    int base = wave * 16;
    if (base >= n_nodes) return;

    int row   = lane & 15;
    int koff  = (lane >> 4) * 8;
    int nrow  = min(base + row, n_nodes - 1);
    const float* ar = agg   + (size_t)nrow * 64;
    const float* hr = nf_in + (size_t)nrow * 64;

    v16bf aA[2], hA[2];
    aA[0] = load_a_frag(ar, 0, koff);  aA[1] = load_a_frag(ar, 32, koff);
    hA[0] = load_a_frag(hr, 0, koff);  hA[1] = load_a_frag(hr, 32, koff);

    int ncol     = lane & 15;
    int mrow_off = (lane < 16) ? 0 : 8;
    int boff     = lane * 8;

    v8f rz[8];
#pragma unroll
    for (int t = 0; t < 8; ++t) {
        int gn = t * 16 + ncol;
        float cinit = bin[gn] + bhn[gn];
        v8f c = zero8();
#pragma unroll
        for (int kt = 0; kt < 2; ++kt)
            c = wmma_bf16(aA[kt], lds_w + NW_WIN + (kt * 12 + t) * 256 + boff, c);
#pragma unroll
        for (int kt = 0; kt < 2; ++kt)
            c = wmma_bf16(hA[kt], lds_w + NW_WHN + (kt * 12 + t) * 256 + boff, c);
#pragma unroll
        for (int i = 0; i < 8; ++i) c[i] = sigmoidf_(c[i] + cinit);
        rz[t] = c;
    }
#pragma unroll
    for (int t = 0; t < 4; ++t) {
        int nt = 8 + t;
        int gn = nt * 16 + ncol;
        float bi0 = bin[gn], bh0 = bhn[gn];
        v8f ci = zero8(), ch = zero8();
#pragma unroll
        for (int kt = 0; kt < 2; ++kt)
            ci = wmma_bf16(aA[kt], lds_w + NW_WIN + (kt * 12 + nt) * 256 + boff, ci);
#pragma unroll
        for (int kt = 0; kt < 2; ++kt)
            ch = wmma_bf16(hA[kt], lds_w + NW_WHN + (kt * 12 + nt) * 256 + boff, ch);
        v8f r = rz[t], z = rz[4 + t];
        int feat = t * 16 + ncol;
#pragma unroll
        for (int i = 0; i < 8; ++i) {
            int n2 = base + i + mrow_off;
            if (n2 < n_nodes) {
                float nn = tanhf_((ci[i] + bi0) + r[i] * (ch[i] + bh0));
                float h  = nf_in[(size_t)n2 * 64 + feat];
                nf_out[(size_t)n2 * 64 + feat] = (1.0f - z[i]) * nn + z[i] * h;
            }
        }
    }
}

// ---------- launch ----------
extern "C" void kernel_launch(void* const* d_in, const int* in_sizes, int n_in,
                              void* d_out, int out_size, void* d_ws, size_t ws_size,
                              hipStream_t stream) {
    const float* nf  = (const float*)d_in[0];
    const float* ef  = (const float*)d_in[1];
    const float* Wie = (const float*)d_in[2];
    const float* Whe = (const float*)d_in[3];
    const float* bie = (const float*)d_in[4];
    const float* bhe = (const float*)d_in[5];
    const float* Win = (const float*)d_in[6];
    const float* Whn = (const float*)d_in[7];
    const float* bin = (const float*)d_in[8];
    const float* bhn = (const float*)d_in[9];
    const float* W1  = (const float*)d_in[10];
    const float* b1  = (const float*)d_in[11];
    const float* W2  = (const float*)d_in[12];
    const float* b2  = (const float*)d_in[13];
    const int*   src = (const int*)d_in[14];
    const int*   dst = (const int*)d_in[15];

    int N = in_sizes[0] / 64;
    int E = in_sizes[14];

    float* out_nf = (float*)d_out;
    float* out_ef = out_nf + (size_t)N * 64;   // uef lives here across both iterations

    // workspace layout: edge weights contiguous, then node weights contiguous
    uint32_t* pw     = (uint32_t*)d_ws;
    uint32_t* pWedge = pw;                       // Wie(12288) | Whe(6144) | W1(6144)
    uint32_t* pWnode = pw + EW_TOT;              // Win(6144) | Whn(6144)
    float*    nf1   = (float*)(pw + EW_TOT + NW_TOT);   // N*64
    float*    logit = nf1 + (size_t)N * 64;             // E (reused as exp buffer)
    unsigned* m_enc = (unsigned*)(logit + E);           // N
    float*    sbuf  = (float*)(m_enc + N);              // N
    float*    agg   = sbuf + N;                         // N*64

    // pack weights into WMMA B-fragment layout (run every call; deterministic)
    {
        int tot;
        tot = (128/32)*12*256; pack_b_kernel<<<(tot+255)/256,256,0,stream>>>(Wie, 128, 12, pWedge + EW_WIE);
        tot = ( 64/32)*12*256; pack_b_kernel<<<(tot+255)/256,256,0,stream>>>(Whe,  64, 12, pWedge + EW_WHE);
        tot = (192/32)* 4*256; pack_b_kernel<<<(tot+255)/256,256,0,stream>>>(W1,  192,  4, pWedge + EW_W1);
        tot = ( 64/32)*12*256; pack_b_kernel<<<(tot+255)/256,256,0,stream>>>(Win,  64, 12, pWnode + NW_WIN);
        tot = ( 64/32)*12*256; pack_b_kernel<<<(tot+255)/256,256,0,stream>>>(Whn,  64, 12, pWnode + NW_WHN);
    }

    int etiles = (E + 15) / 16, eblocks = (etiles + 7) / 8;
    int ntiles = (N + 15) / 16, nblocks = (ntiles + 7) / 8;
    int initg  = (N * 64 + 255) / 256;
    int aggblk = (int)(((long long)E * 64 + 255) / 256);

    const float* nf_in = nf;
    const float* ef_in = ef;
    for (int it = 0; it < 2; ++it) {
        init_seg_kernel<<<initg, 256, 0, stream>>>(m_enc, sbuf, agg, N);
        edge_kernel<<<eblocks, 256, EW_TOT * 4, stream>>>(nf_in, ef_in, src, dst,
                                                          pWedge, bie, bhe, b1, W2, b2,
                                                          out_ef, logit, m_enc, E);
        exp_sum_kernel<<<(E + 255) / 256, 256, 0, stream>>>(logit, m_enc, dst, sbuf, E);
        agg_scatter_kernel<<<aggblk, 256, 0, stream>>>(out_ef, logit, sbuf, dst, agg, E);
        node_kernel<<<nblocks, 256, NW_TOT * 4, stream>>>(agg, nf_in, pWnode, bin, bhn,
                                                          (it == 0) ? nf1 : out_nf, N);
        nf_in = nf1;
        ef_in = out_ef;   // iteration 2 updates edge features in place (tile-local rows)
    }
}
